// HEPOSCrossAttention_24730421690610
// MI455X (gfx1250) — matrix operand order repaired and checked by reading, not compiled, not run on hardware
//
#include <hip/hip_runtime.h>

// ---------------------------------------------------------------------------
// HEPOS cross-attention for MI455X (gfx1250, wave32, WMMA bf16 16x16x32)
// ---------------------------------------------------------------------------

typedef __attribute__((ext_vector_type(16))) __bf16          v16bf;
typedef __attribute__((ext_vector_type(16))) unsigned short  v16u;
typedef __attribute__((ext_vector_type(8)))  float           v8f;
typedef __attribute__((ext_vector_type(8)))  unsigned short  u16x8;
typedef __attribute__((ext_vector_type(4)))  unsigned short  u16x4;

#define D_MODEL 1024
#define NHEADS  16
#define HDIM    64
#define STRIDE  4
#define B_      2
#define T_      512
#define S_      8192
#define SEFF    2048          // S / STRIDE (effective keys per head)
#define SCALE_F 0.125f        // HEAD_DIM^-0.5
#define NEG_BIG -1.0e30f

__device__ __forceinline__ unsigned short f2bf(float f) {
    unsigned int u = __float_as_uint(f);
    return (unsigned short)((u + 0x7FFFu + ((u >> 16) & 1u)) >> 16);  // RNE
}

// concat two 8-element halves into one 16-element WMMA operand
__device__ __forceinline__ v16bf concat8(u16x8 lo, u16x8 hi) {
    v16u t = __builtin_shufflevector(lo, hi, 0, 1, 2, 3, 4, 5, 6, 7,
                                     8, 9, 10, 11, 12, 13, 14, 15);
    return __builtin_bit_cast(v16bf, t);
}

// contiguous 32-byte fragment load (2x b128 straight into place, no moves)
__device__ __forceinline__ v16bf ldfrag(const unsigned short* p) {
    return __builtin_bit_cast(v16bf, *(const v16u*)p);
}

__device__ __forceinline__ v8f vzero8() {
    v8f z;
#pragma unroll
    for (int i = 0; i < 8; i++) z[i] = 0.0f;
    return z;
}

// ---------------------------------------------------------------------------
// Tiled GEMM: C[64x64 tile] = A(f32) * W(f32) + bias, bf16 WMMA inner loop.
// MODE 0: Q proj   M=1024 N=1024 -> qh[b][h][t][64]            (bf16)
// MODE 1: K proj   M=4096 N=256  (group g) -> kh[b][h][j][64]  (bf16)
// MODE 2: V proj   M=4096 N=256  (group g) -> vt[b][h][d][2048](bf16, transposed)
// MODE 3: out proj M=1024 N=1024 -> d_out f32
// Block: 128 threads = 4 waves; wave w owns rows w*16..w*16+15 of the tile.
// ---------------------------------------------------------------------------
template <int MODE>
__global__ __launch_bounds__(128) void gemm_kernel(const float* __restrict__ A,
                                                   const float* __restrict__ W,
                                                   const float* __restrict__ bias,
                                                   void* __restrict__ outp) {
    __shared__ alignas(32) unsigned short As[64][48];   // [m][k] k=32 (+16 pad)
    __shared__ alignas(32) unsigned short Bt[64][48];   // [n][k] transposed

    const int tid  = threadIdx.x;
    const int lane = tid & 31;
    const int wave = tid >> 5;
    const int n16  = lane & 15;
    const int half = lane >> 4;
    const int g    = (MODE == 1 || MODE == 2) ? blockIdx.z : 0;
    const int blockM = blockIdx.y * 64;
    const int blockN = blockIdx.x * 64;

    // cooperative-load coordinates
    const int ar = tid >> 1;   // A tile row 0..63
    const int ah = tid & 1;    // which 16-float half of the 32-wide k slab
    const int kq = tid >> 4;   // W sub-block: k rows 4*kq..4*kq+3
    const int nq = tid & 15;   // W sub-block: n cols 4*nq..4*nq+3

    // A global row (with HEPOS row gather for K/V)
    const int mrowA = blockM + ar;
    size_t arow;
    if constexpr (MODE == 1 || MODE == 2) {
        const int bb = mrowA >> 11, j = mrowA & 2047;
        arow = (size_t)bb * S_ + (size_t)(g + STRIDE * j);
    } else {
        arow = (size_t)mrowA;
    }
    const float* aptr = A + arow * D_MODEL + ah * 16;

    // W column base for this thread's 4-col group (HEPOS column gather for K/V)
    const int ncol0 = blockN + nq * 4;
    int wcol0;
    if constexpr (MODE == 1 || MODE == 2)
        wcol0 = g * HDIM + (ncol0 >> 6) * (STRIDE * HDIM) + (ncol0 & 63);
    else
        wcol0 = ncol0;

    v8f acc[4];
#pragma unroll
    for (int t = 0; t < 4; t++) acc[t] = vzero8();

    for (int k0 = 0; k0 < D_MODEL; k0 += 32) {
        if (k0 + 32 < D_MODEL) {
            __builtin_prefetch(aptr + k0 + 32, 0, 3);
            __builtin_prefetch(W + (size_t)(k0 + 32 + kq * 4) * D_MODEL + wcol0, 0, 3);
        }
        // --- stage A slab (64x32 f32 -> bf16), one 32B vector store ---
        {
            const float4* a4 = (const float4*)(aptr + k0);
            float4 av0 = a4[0], av1 = a4[1], av2 = a4[2], av3 = a4[3];
            v16u av;
            av[0]  = f2bf(av0.x); av[1]  = f2bf(av0.y); av[2]  = f2bf(av0.z); av[3]  = f2bf(av0.w);
            av[4]  = f2bf(av1.x); av[5]  = f2bf(av1.y); av[6]  = f2bf(av1.z); av[7]  = f2bf(av1.w);
            av[8]  = f2bf(av2.x); av[9]  = f2bf(av2.y); av[10] = f2bf(av2.z); av[11] = f2bf(av2.w);
            av[12] = f2bf(av3.x); av[13] = f2bf(av3.y); av[14] = f2bf(av3.z); av[15] = f2bf(av3.w);
            *(v16u*)&As[ar][ah * 16] = av;
        }
        // --- stage W slab (32x64 f32 -> bf16, transposed): 4k x 4n sub-block,
        //     stores become 4x ds_store_b64 along k ---
        {
            float4 wv[4];
#pragma unroll
            for (int kk = 0; kk < 4; kk++)
                wv[kk] = *(const float4*)(W + (size_t)(k0 + kq * 4 + kk) * D_MODEL + wcol0);
#pragma unroll
            for (int j = 0; j < 4; j++) {
                u16x4 col;
                col[0] = f2bf(((const float*)&wv[0])[j]);
                col[1] = f2bf(((const float*)&wv[1])[j]);
                col[2] = f2bf(((const float*)&wv[2])[j]);
                col[3] = f2bf(((const float*)&wv[3])[j]);
                *(u16x4*)&Bt[nq * 4 + j][kq * 4] = col;
            }
        }
        __syncthreads();

        // --- fragments + 4 WMMAs per wave ---
        const int ml = wave * 16 + n16;
        u16x8 ra = *(const u16x8*)&As[ml][half * 16];
        u16x8 rb = *(const u16x8*)&As[ml][half * 16 + 8];
        v16bf af = concat8(ra, rb);
#pragma unroll
        for (int t = 0; t < 4; t++) {
            v16bf bf = ldfrag(&Bt[t * 16 + n16][half * 16]);
            acc[t] = __builtin_amdgcn_wmma_f32_16x16x32_bf16(
                false, af, false, bf, (short)0, acc[t], false, false);
        }
        __syncthreads();
    }

    // --- epilogue: bias + layout-specific store (C layout: row = half*8+r) ---
#pragma unroll
    for (int t = 0; t < 4; t++) {
        const int gcol = blockN + t * 16 + n16;
        int bcol;
        if constexpr (MODE == 1 || MODE == 2)
            bcol = g * HDIM + (gcol >> 6) * (STRIDE * HDIM) + (gcol & 63);
        else
            bcol = gcol;
        const float bv = bias[bcol];
#pragma unroll
        for (int r = 0; r < 8; r++) {
            const int grow = blockM + wave * 16 + half * 8 + r;
            const float val = acc[t][r] + bv;
            if constexpr (MODE == 0) {
                const int bb = grow >> 9, tt = grow & 511;
                const int h = gcol >> 6, d = gcol & 63;
                ((unsigned short*)outp)[(((size_t)(bb * NHEADS + h) * T_ + tt) * HDIM) + d] = f2bf(val);
            } else if constexpr (MODE == 1) {
                const int bb = grow >> 11, j = grow & 2047;
                const int h = g + STRIDE * (gcol >> 6), d = gcol & 63;
                ((unsigned short*)outp)[(((size_t)(bb * NHEADS + h) * SEFF + j) * HDIM) + d] = f2bf(val);
            } else if constexpr (MODE == 2) {
                const int bb = grow >> 11, j = grow & 2047;
                const int h = g + STRIDE * (gcol >> 6), d = gcol & 63;
                ((unsigned short*)outp)[(((size_t)(bb * NHEADS + h) * HDIM + d) * SEFF) + j] = f2bf(val);
            } else {
                ((float*)outp)[(size_t)grow * D_MODEL + gcol] = val;
            }
        }
    }
}

// ---------------------------------------------------------------------------
// Flash-style attention: one wave per (b, h, 16-row Q tile).
// 64 chunks of 32 effective keys; per chunk: 4 score WMMAs + 5 context WMMAs
// (5th tile = all-ones B matrix, computes the softmax row-sum on the matrix
//  pipe instead of 32 ds_bpermute shuffles; online rescale comes for free).
// ---------------------------------------------------------------------------
__global__ __launch_bounds__(128) void attn_kernel(const unsigned short* __restrict__ qh,
                                                   const unsigned short* __restrict__ kh,
                                                   const unsigned short* __restrict__ vt,
                                                   const unsigned char* __restrict__ kpm,
                                                   float* __restrict__ ctx) {
    __shared__ alignas(16) unsigned short P[4][16][48];  // per-wave 16x32 P tile

    const int tid  = threadIdx.x;
    const int lane = tid & 31;
    const int wave = tid >> 5;
    const int n16  = lane & 15;
    const int half = lane >> 4;

    const int task  = blockIdx.x * 4 + wave;  // 0..1023
    const int ttile = task & 31;
    const int h     = (task >> 5) & 15;
    const int b     = task >> 9;
    const int g     = h & 3;

    // Q fragments (A layout: row m = lane&15, K split by lane half)
    const unsigned short* qrow = qh + (((size_t)(b * NHEADS + h) * T_ + ttile * 16 + n16) * HDIM);
    v16bf qa[2];
#pragma unroll
    for (int u = 0; u < 2; u++) {
        u16x8 raq = *(const u16x8*)(qrow + u * 32 + half * 16);
        u16x8 rbq = *(const u16x8*)(qrow + u * 32 + half * 16 + 8);
        qa[u] = concat8(raq, rbq);
    }

    // all-ones bf16 B fragment (bf16 1.0 = 0x3F80)
    v16u onesu;
#pragma unroll
    for (int i = 0; i < 16; i++) onesu[i] = 0x3F80u;
    const v16bf onesf = __builtin_bit_cast(v16bf, onesu);

    const unsigned short* kbase = kh + (size_t)(b * NHEADS + h) * SEFF * HDIM;
    const unsigned short* vbase = vt + (size_t)(b * NHEADS + h) * HDIM * SEFF;
    const unsigned char*  mbase = kpm + (size_t)b * S_;

    float mx[8];
#pragma unroll
    for (int r = 0; r < 8; r++) mx[r] = NEG_BIG;
    v8f co[5];  // 4 context tiles + row-sum tile
#pragma unroll
    for (int t = 0; t < 5; t++) co[t] = vzero8();

    for (int c0 = 0; c0 < SEFF; c0 += 32) {
        if (c0 + 32 < SEFF) {
            __builtin_prefetch(kbase + (size_t)(c0 + 32 + lane) * HDIM, 0, 3);
            __builtin_prefetch(vbase + (size_t)lane * SEFF + c0 + 32, 0, 3);
            __builtin_prefetch(vbase + (size_t)(lane + 32) * SEFF + c0 + 32, 0, 3);
        }
        // ---- scores: S[16x32] = Q[16x64] . K^T (contiguous 32B B-frags) ----
        const unsigned short* k0p = kbase + (size_t)(c0 + n16) * HDIM + half * 16;
        const unsigned short* k1p = kbase + (size_t)(c0 + 16 + n16) * HDIM + half * 16;
        v8f sc0 = vzero8(), sc1 = vzero8();
        sc0 = __builtin_amdgcn_wmma_f32_16x16x32_bf16(false, qa[0], false, ldfrag(k0p),      (short)0, sc0, false, false);
        sc0 = __builtin_amdgcn_wmma_f32_16x16x32_bf16(false, qa[1], false, ldfrag(k0p + 32), (short)0, sc0, false, false);
        sc1 = __builtin_amdgcn_wmma_f32_16x16x32_bf16(false, qa[0], false, ldfrag(k1p),      (short)0, sc1, false, false);
        sc1 = __builtin_amdgcn_wmma_f32_16x16x32_bf16(false, qa[1], false, ldfrag(k1p + 32), (short)0, sc1, false, false);

        // key padding mask (column-wise, raw s = g + 4*key)
        const float bias0 = mbase[g + STRIDE * (c0 + n16)]      ? NEG_BIG : 0.0f;
        const float bias1 = mbase[g + STRIDE * (c0 + 16 + n16)] ? NEG_BIG : 0.0f;

        // ---- online softmax (rows live across 16-lane halves) ----
#pragma unroll
        for (int r = 0; r < 8; r++) {
            float s0 = sc0[r] * SCALE_F + bias0;
            float s1 = sc1[r] * SCALE_F + bias1;
            float cm = fmaxf(s0, s1);
            cm = fmaxf(cm, __shfl_xor(cm, 1));
            cm = fmaxf(cm, __shfl_xor(cm, 2));
            cm = fmaxf(cm, __shfl_xor(cm, 4));
            cm = fmaxf(cm, __shfl_xor(cm, 8));
            const float mn = fmaxf(mx[r], cm);
            // clamp form: fmaxf absorbs -inf/NaN, no compares needed
            const float sc_old = __expf(fmaxf(mx[r] - mn, -80.0f));
            const float p0     = __expf(fmaxf(s0 - mn, -80.0f));
            const float p1     = __expf(fmaxf(s1 - mn, -80.0f));
            mx[r] = mn;
#pragma unroll
            for (int t4 = 0; t4 < 5; t4++) co[t4][r] *= sc_old;
            P[wave][half * 8 + r][n16]      = f2bf(p0);
            P[wave][half * 8 + r][16 + n16] = f2bf(p1);
        }
        // cross-lane C-layout -> A-layout through LDS: drain DS stores
        asm volatile("s_wait_dscnt 0" ::: "memory");
        u16x8 pa0 = *(const u16x8*)&P[wave][n16][half * 16];
        u16x8 pa1 = *(const u16x8*)&P[wave][n16][half * 16 + 8];
        v16bf pf = concat8(pa0, pa1);
        asm volatile("" ::: "memory");  // keep next-iter stores after these loads

        // ---- context: C[16x64] += P[16x32] . V[32x64] (V stored transposed),
        //      tile 4 = ones -> running softmax denominator ----
#pragma unroll
        for (int t4 = 0; t4 < 4; t4++) {
            const unsigned short* vrow = vbase + (size_t)(t4 * 16 + n16) * SEFF + c0 + half * 16;
            co[t4] = __builtin_amdgcn_wmma_f32_16x16x32_bf16(
                false, pf, false, ldfrag(vrow), (short)0, co[t4], false, false);
        }
        co[4] = __builtin_amdgcn_wmma_f32_16x16x32_bf16(
            false, pf, false, onesf, (short)0, co[4], false, false);
    }

    // ---- normalize + store context (f32, [B,T,1024] with head columns) ----
#pragma unroll
    for (int t4 = 0; t4 < 4; t4++) {
#pragma unroll
        for (int r = 0; r < 8; r++) {
            const int row  = ttile * 16 + half * 8 + r;
            const float ls = co[4][r];  // row sum, identical across the 16-lane half
            const float inv = (ls > 1e-30f) ? (1.0f / ls) : 0.0f;
            ctx[((size_t)b * T_ + row) * D_MODEL + h * HDIM + t4 * 16 + n16] = co[t4][r] * inv;
        }
    }
}

// ---------------------------------------------------------------------------
extern "C" void kernel_launch(void* const* d_in, const int* in_sizes, int n_in,
                              void* d_out, int out_size, void* d_ws, size_t ws_size,
                              hipStream_t stream) {
    (void)in_sizes; (void)n_in; (void)out_size; (void)ws_size;

    const float*         query = (const float*)d_in[0];
    const float*         key   = (const float*)d_in[1];
    const float*         value = (const float*)d_in[2];
    const unsigned char* kpm   = (const unsigned char*)d_in[3];
    const float*         Wq    = (const float*)d_in[4];
    const float*         bq    = (const float*)d_in[5];
    const float*         Wk    = (const float*)d_in[6];
    const float*         bk    = (const float*)d_in[7];
    const float*         Wv    = (const float*)d_in[8];
    const float*         bv    = (const float*)d_in[9];
    const float*         Wo    = (const float*)d_in[10];
    const float*         bo    = (const float*)d_in[11];

    char* ws = (char*)d_ws;
    unsigned short* qh  = (unsigned short*)(ws);                      //  2 MB
    unsigned short* kh  = (unsigned short*)(ws + (size_t)(2  << 20)); //  8 MB
    unsigned short* vtb = (unsigned short*)(ws + (size_t)(10 << 20)); //  8 MB
    float*          ctx = (float*)(ws + (size_t)(18 << 20));          //  4 MB

    // Q projection (full), K/V projections (HEPOS group-compacted, 4x less work)
    gemm_kernel<0><<<dim3(16, 16, 1), 128, 0, stream>>>(query, Wq, bq, qh);
    gemm_kernel<1><<<dim3(4, 64, 4),  128, 0, stream>>>(key,   Wk, bk, kh);
    gemm_kernel<2><<<dim3(4, 64, 4),  128, 0, stream>>>(value, Wv, bv, vtb);
    // flash attention over 2048 effective keys per head
    attn_kernel<<<dim3(256, 1, 1), 128, 0, stream>>>(qh, kh, vtb, kpm, ctx);
    // output projection
    gemm_kernel<3><<<dim3(16, 16, 1), 128, 0, stream>>>(ctx, Wo, bo, d_out);
}
